// SingleSideCELoss_79328045957202
// MI455X (gfx1250) — matrix operand order repaired and checked by reading, not compiled, not run on hardware
//
#include <hip/hip_runtime.h>
#include <hip/hip_bf16.h>
#include <math.h>

// Problem shape (fixed by the reference).
#define N_ROWS 16384
#define P_COLS 1024
#define D_DIM  512
#define MBLK   32                 // rows per workgroup = 2 M-tiles of 16
#define KSTEPS (D_DIM / 32)

typedef __attribute__((ext_vector_type(16))) __bf16 v16bf;
typedef __attribute__((ext_vector_type(8)))  float  v8f;

// ---------- bf16 helpers (RNE, bit-exact) -----------------------------------
__device__ __forceinline__ unsigned short f32_to_bf16_rne(float f) {
    unsigned u = __float_as_uint(f);
    u += 0x7FFFu + ((u >> 16) & 1u);
    return (unsigned short)(u >> 16);
}
__device__ __forceinline__ float bf16_to_f32(unsigned short h) {
    return __uint_as_float(((unsigned)h) << 16);
}

// ---------- pass 0a: split f32 -> bf16 hi + bf16 lo residual ----------------
__global__ void convert_split_kernel(const float* __restrict__ src,
                                     unsigned short* __restrict__ hi,
                                     unsigned short* __restrict__ lo,
                                     int n) {
    int i = blockIdx.x * blockDim.x + threadIdx.x;
    if (i >= n) return;
    float x = src[i];
    unsigned short h = f32_to_bf16_rne(x);
    float r = x - bf16_to_f32(h);
    hi[i] = h;
    lo[i] = f32_to_bf16_rne(r);
}

// ---------- pass 0b: per-row sum of squares (one wave32 per row) ------------
__global__ void row_sumsq_kernel(const float* __restrict__ x,
                                 float* __restrict__ out,
                                 int rows, int cols) {
    int wave = threadIdx.x >> 5;
    int lane = threadIdx.x & 31;
    int row  = blockIdx.x * (blockDim.x >> 5) + wave;
    if (row >= rows) return;
    const float* p = x + (size_t)row * cols;
    float s = 0.f;
    for (int c = lane; c < cols; c += 32) { float v = p[c]; s += v * v; }
    #pragma unroll
    for (int off = 16; off; off >>= 1) s += __shfl_xor(s, off, 32);
    if (lane == 0) out[row] = s;
}

// ---------- WMMA fragment assembly: two contiguous 16B loads per operand ----
union FragU {
    unsigned short us[16];
    v16bf          v;
    float4         f4[2];
};
__device__ __forceinline__ v16bf load_frag16(const unsigned short* p0,
                                             const unsigned short* p1) {
    FragU u;
    u.f4[0] = *reinterpret_cast<const float4*>(p0);
    u.f4[1] = *reinterpret_cast<const float4*>(p1);
    return u.v;
}

// ---------- pass 1: fused WMMA GEMM + distance + softmax + NLL --------------
// Grid: N/32 blocks of 256 threads (8 waves). Each block: 32 feature rows vs
// all 1024 prototypes. Wave w owns columns [w*128, w*128+128) for BOTH
// 16-row M-tiles (128 accumulator VGPRs) -> each B fragment feeds 6 WMMAs.
// The 32x512 hi/lo A panel is async-staged to LDS once and shared by all waves.
__global__ __launch_bounds__(256, 1)
void ce_wmma_kernel(const unsigned short* __restrict__ fhi,
                    const unsigned short* __restrict__ flo,
                    const unsigned short* __restrict__ phi,
                    const unsigned short* __restrict__ plo,
                    const float* __restrict__ f2,
                    const float* __restrict__ p2,
                    const int* __restrict__ domain,
                    float* __restrict__ partials) {
    __shared__ __attribute__((aligned(16))) unsigned short sAhi[MBLK * D_DIM]; // 32 KB
    __shared__ __attribute__((aligned(16))) unsigned short sAlo[MBLK * D_DIM]; // 32 KB
    __shared__ float s_red[8][MBLK];   // cross-wave reduction scratch
    __shared__ float s_gmax[MBLK];
    __shared__ float s_simt[MBLK];     // sim at target class, per row
    __shared__ float s_f2[MBLK];
    __shared__ int   s_tcls[MBLK];

    const int tid  = threadIdx.x;
    const int wave = tid >> 5;
    const int lane = tid & 31;
    const int half = lane >> 4;        // lane group 0-15 vs 16-31
    const int l15  = lane & 15;
    const int rb   = blockIdx.x;       // 32-row block index

    if (tid < MBLK) {
        const int g = rb * MBLK + tid;
        s_f2[tid]   = f2[g];
        s_tcls[tid] = domain[g] & (P_COLS - 1);
    }

    // ---- async-stage the 32 x 512 A panel (hi+lo) into LDS -----------------
    {
        const unsigned dhib = (unsigned)(size_t)&sAhi[0];
        const unsigned dlob = (unsigned)(size_t)&sAlo[0];
        const size_t gbase = (size_t)rb * MBLK * D_DIM;
        #pragma unroll
        for (int it = 0; it < (MBLK * D_DIM / 8) / 256; ++it) {
            const int c = it * 256 + tid;               // 16-byte chunk id
            const size_t goff = gbase + (size_t)c * 8;  // ushort offset
            unsigned long long ghi = (unsigned long long)(fhi + goff);
            unsigned long long glo = (unsigned long long)(flo + goff);
            unsigned dhi = dhib + (unsigned)c * 16u;
            unsigned dlo = dlob + (unsigned)c * 16u;
            asm volatile("global_load_async_to_lds_b128 %0, %1, off"
                         :: "v"(dhi), "v"(ghi) : "memory");
            asm volatile("global_load_async_to_lds_b128 %0, %1, off"
                         :: "v"(dlo), "v"(glo) : "memory");
        }
        asm volatile("s_wait_asynccnt 0x0" ::: "memory");
    }
    __syncthreads();

    v8f zero = {};
    v8f acc[2][8];
    #pragma unroll
    for (int mt = 0; mt < 2; ++mt)
        #pragma unroll
        for (int t = 0; t < 8; ++t) acc[mt][t] = zero;

    #pragma unroll 1
    for (int ks = 0; ks < KSTEPS; ++ks) {
        const int k0 = ks * 32;
        // A 16x32 bf16 layout: lane<16 -> K[k0..k0+7],K[k0+16..k0+23];
        // lanes 16-31 shifted by 8. Two 16B LDS loads per fragment.
        v16bf ah[2], al[2];
        #pragma unroll
        for (int mt = 0; mt < 2; ++mt) {
            const unsigned aoff =
                (unsigned)((mt * 16 + l15) * D_DIM + k0 + half * 8);
            ah[mt] = load_frag16(sAhi + aoff, sAhi + aoff + 16);
            al[mt] = load_frag16(sAlo + aoff, sAlo + aoff + 16);
        }
        #pragma unroll
        for (int t = 0; t < 8; ++t) {
            const int col = wave * 128 + t * 16 + l15;   // prototype row
            const size_t boff =
                (size_t)col * D_DIM + (size_t)k0 + (size_t)half * 8;
            if (t == 0 && ks + 1 < KSTEPS)
                __builtin_prefetch(phi + boff + 32, 0, 3);  // global_prefetch_b8
            v16bf bh = load_frag16(phi + boff, phi + boff + 16);
            v16bf bl = load_frag16(plo + boff, plo + boff + 16);
            #pragma unroll
            for (int mt = 0; mt < 2; ++mt) {
                // bf16x3 split: hi*hi + hi*lo + lo*hi, f32 accumulate
                acc[mt][t] = __builtin_amdgcn_wmma_f32_16x16x32_bf16(
                    false, ah[mt], false, bh, (short)0, acc[mt][t], false, false);
                acc[mt][t] = __builtin_amdgcn_wmma_f32_16x16x32_bf16(
                    false, ah[mt], false, bl, (short)0, acc[mt][t], false, false);
                acc[mt][t] = __builtin_amdgcn_wmma_f32_16x16x32_bf16(
                    false, al[mt], false, bh, (short)0, acc[mt][t], false, false);
            }
        }
    }

    // ---- epilogue, register-resident --------------------------------------
    // C/D layout: VGPR j, lanes 0-15 -> M=j, lanes 16-31 -> M=j+8; N=lane%16.
    // 1) sim = -sqrt(max(f2+p2-2*cross,0)) in place; extract target logits.
    #pragma unroll
    for (int mt = 0; mt < 2; ++mt)
    #pragma unroll
    for (int t = 0; t < 8; ++t) {
        const int col = wave * 128 + t * 16 + l15;
        const float p2v = p2[col];
        #pragma unroll
        for (int j = 0; j < 8; ++j) {
            const int r = mt * 16 + half * 8 + j;
            const float cr = acc[mt][t][j];
            const float sq = fmaxf(s_f2[r] + p2v - 2.0f * cr, 0.0f);
            const float sv = -sqrtf(sq);
            acc[mt][t][j] = sv;
            if (col == s_tcls[r]) s_simt[r] = sv;   // exactly one writer per row
        }
    }

    // 2) per-row max: lane-local over tiles, xor-shuffle over 16-lane group.
    #pragma unroll
    for (int mt = 0; mt < 2; ++mt)
    #pragma unroll
    for (int j = 0; j < 8; ++j) {
        float m = -INFINITY;
        #pragma unroll
        for (int t = 0; t < 8; ++t) m = fmaxf(m, acc[mt][t][j]);
        #pragma unroll
        for (int off = 1; off < 16; off <<= 1)
            m = fmaxf(m, __shfl_xor(m, off, 32));
        if (l15 == j) s_red[wave][mt * 16 + half * 8 + j] = m;
    }
    __syncthreads();
    if (tid < MBLK) {
        float m = -INFINITY;
        #pragma unroll
        for (int w = 0; w < 8; ++w) m = fmaxf(m, s_red[w][tid]);
        s_gmax[tid] = m;
    }
    __syncthreads();

    // 3) per-row sum of exp, same reduction pattern (fixed order).
    #pragma unroll
    for (int mt = 0; mt < 2; ++mt)
    #pragma unroll
    for (int j = 0; j < 8; ++j) {
        const int r = mt * 16 + half * 8 + j;
        const float gm = s_gmax[r];
        float s = 0.0f;
        #pragma unroll
        for (int t = 0; t < 8; ++t) s += expf(acc[mt][t][j] - gm);
        #pragma unroll
        for (int off = 1; off < 16; off <<= 1) s += __shfl_xor(s, off, 32);
        if (l15 == j) s_red[wave][r] = s;
    }
    __syncthreads();

    // 4) NLL per row, block sum via wave-0 shuffle (deterministic).
    if (tid < 32) {
        float s = 0.0f;
        #pragma unroll
        for (int w = 0; w < 8; ++w) s += s_red[w][tid];
        float nll = -(s_simt[tid] - s_gmax[tid] - logf(s));
        #pragma unroll
        for (int off = 16; off; off >>= 1) nll += __shfl_xor(nll, off, 32);
        if (lane == 0) partials[rb] = nll;
    }
}

// ---------- pass 2: fixed-order deterministic reduction ---------------------
__global__ void final_reduce_kernel(const float* __restrict__ partials,
                                    float* __restrict__ out,
                                    int n, float scale) {
    __shared__ float buf[256];
    float s = 0.0f;
    for (int i = threadIdx.x; i < n; i += 256) s += partials[i];
    buf[threadIdx.x] = s;
    __syncthreads();
    #pragma unroll
    for (int stride = 128; stride; stride >>= 1) {
        if (threadIdx.x < stride) buf[threadIdx.x] += buf[threadIdx.x + stride];
        __syncthreads();
    }
    if (threadIdx.x == 0) out[0] = buf[0] * scale;
}

// ---------- launcher --------------------------------------------------------
extern "C" void kernel_launch(void* const* d_in, const int* in_sizes, int n_in,
                              void* d_out, int out_size, void* d_ws, size_t ws_size,
                              hipStream_t stream) {
    const float* feat  = (const float*)d_in[0];   // [N, D] f32
    const float* proto = (const float*)d_in[1];   // [P, D] f32
    const int*   dom   = (const int*)d_in[2];     // [N] int

    char* ws = (char*)d_ws;
    size_t off = 0;
    auto wsalloc = [&](size_t bytes) -> void* {
        void* p = ws + off;
        off += (bytes + 255) & ~(size_t)255;
        return p;
    };
    unsigned short* fhi = (unsigned short*)wsalloc((size_t)N_ROWS * D_DIM * 2);
    unsigned short* flo = (unsigned short*)wsalloc((size_t)N_ROWS * D_DIM * 2);
    unsigned short* phi = (unsigned short*)wsalloc((size_t)P_COLS * D_DIM * 2);
    unsigned short* plo = (unsigned short*)wsalloc((size_t)P_COLS * D_DIM * 2);
    float* f2       = (float*)wsalloc((size_t)N_ROWS * 4);
    float* p2       = (float*)wsalloc((size_t)P_COLS * 4);
    float* partials = (float*)wsalloc((size_t)(N_ROWS / MBLK) * 4);

    const int nf = N_ROWS * D_DIM;
    const int np = P_COLS * D_DIM;
    convert_split_kernel<<<(nf + 255) / 256, 256, 0, stream>>>(feat, fhi, flo, nf);
    convert_split_kernel<<<(np + 255) / 256, 256, 0, stream>>>(proto, phi, plo, np);
    row_sumsq_kernel<<<(N_ROWS + 7) / 8, 256, 0, stream>>>(feat, f2, N_ROWS, D_DIM);
    row_sumsq_kernel<<<(P_COLS + 7) / 8, 256, 0, stream>>>(proto, p2, P_COLS, D_DIM);

    ce_wmma_kernel<<<N_ROWS / MBLK, 256, 0, stream>>>(fhi, flo, phi, plo,
                                                      f2, p2, dom, partials);

    final_reduce_kernel<<<1, 256, 0, stream>>>(partials, (float*)d_out,
                                               N_ROWS / MBLK,
                                               1.0f / (float)N_ROWS);
}
